// ResidualSAGENet_41506563948594
// MI455X (gfx1250) — matrix operand order repaired and checked by reading, not compiled, not run on hardware
//
#include <hip/hip_runtime.h>
#include <hip/hip_bf16.h>

// ---------------------------------------------------------------------------
// ResidualSAGENet on gfx1250: bf16 WMMA GEMMs + f32 hw-atomic aggregation.
// N=100000 nodes (6250 x 16-row tiles, exact), E=600000 edges, H=128, C=47.
// ---------------------------------------------------------------------------

#define GN 100000
#define GE 600000
#define GH 128
#define GC 47
#define GL 3
#define LDST 136          // LDS row stride in bf16 elems (128 + 8 pad = 272B)

typedef __attribute__((ext_vector_type(16))) __bf16 v16bf;
typedef __attribute__((ext_vector_type(8)))  float  v8f;

__device__ __forceinline__ unsigned short f2bf(float f) {
    unsigned int u = __float_as_uint(f);
    unsigned int r = (u + 0x7FFFu + ((u >> 16) & 1u)) >> 16;  // RNE
    return (unsigned short)r;
}

// Load one 16x32 bf16 A-fragment from LDS (ISA 16-bit A layout):
// lane 0-15: M=lane, K = kt*32 + {0..7, 16..23}
// lane 16-31: M=lane-16, K = kt*32 + {8..15, 24..31}
__device__ __forceinline__ v16bf load_a_frag(const unsigned short* s, int lane, int kt) {
    int mrow = lane & 15;
    int off  = kt * 32 + ((lane >> 4) << 3);
    const unsigned short* p = s + mrow * LDST + off;
    union { float4 f[2]; v16bf v; } U;
    U.f[0] = *(const float4*)(p);        // K group 0..7
    U.f[1] = *(const float4*)(p + 16);   // K group 16..23
    return U.v;
}

// ---------------------------------------------------------------------------
// Weight packing: row-major [128][128] f32 -> B-fragment order bf16.
// out[kt][nt][lane][slot]; lane<16: n=nt*16+lane, K=kt*32+{0..7,16..23}
//                          lane>=16: K=kt*32+{8..15,24..31}
// ---------------------------------------------------------------------------
__global__ void pack128_kernel(const float* __restrict__ W,
                               unsigned short* __restrict__ out) {
    int t    = blockIdx.x * 256 + threadIdx.x;      // 0..16383
    int slot = t & 15;
    int lane = (t >> 4) & 31;
    int nt   = (t >> 9) & 7;
    int kt   = t >> 12;
    int k = kt * 32 + ((lane >> 4) << 3) + (slot & 7) + ((slot >> 3) << 4);
    int n = nt * 16 + (lane & 15);
    out[t] = f2bf(W[k * GH + n]);
}

__global__ void packout_kernel(const float* __restrict__ W,   // [128][47]
                               unsigned short* __restrict__ out) {
    int t    = blockIdx.x * 256 + threadIdx.x;      // 0..6143 (4 kt x 3 nt)
    int slot = t & 15;
    int lane = (t >> 4) & 31;
    int g    = t >> 9;            // 0..11
    int nt   = g % 3;
    int kt   = g / 3;
    int k = kt * 32 + ((lane >> 4) << 3) + (slot & 7) + ((slot >> 3) << 4);
    int n = nt * 16 + (lane & 15);
    out[t] = (n < GC) ? f2bf(W[k * GC + n]) : (unsigned short)0;
}

// ---------------------------------------------------------------------------
// Degree (float) per destination node.
// ---------------------------------------------------------------------------
__global__ void degree_kernel(const int* __restrict__ dst,
                              float* __restrict__ deg, int E) {
    int e = blockIdx.x * blockDim.x + threadIdx.x;
    if (e < E) unsafeAtomicAdd(&deg[dst[e]], 1.0f);
}

// ---------------------------------------------------------------------------
// Edge scatter: agg[dst] += h[src].  One wave per edge, float4 per lane,
// hardware global_atomic_add_f32.  h and agg both resident in 192MB L2.
// ---------------------------------------------------------------------------
__global__ __launch_bounds__(256) void scatter_kernel(
    const float* __restrict__ H, const int* __restrict__ src,
    const int* __restrict__ dst, float* __restrict__ agg, int E) {
    int w = blockIdx.x * 8 + (threadIdx.x >> 5);   // wave-uniform edge id
    if (w >= E) return;
    int lane = threadIdx.x & 31;
    int s = src[w];
    int d = dst[w];
    float4 v = *(const float4*)(H + (size_t)s * GH + lane * 4);
    float* p = agg + (size_t)d * GH + lane * 4;
    unsafeAtomicAdd(p + 0, v.x);
    unsafeAtomicAdd(p + 1, v.y);
    unsafeAtomicAdd(p + 2, v.z);
    unsafeAtomicAdd(p + 3, v.w);
}

// ---------------------------------------------------------------------------
// Fused (SAGE) layer:  Hout = [RES? X:0] + relu(bn(X@Wr + agg_mean@Wl + bias))
// Block = 256 thr = 8 waves = one 16-row x 128-col tile.  Wave w owns ntile w.
// Compile-time HAS_AGG/RES -> branch-free k-loop with two independent WMMA
// accumulation chains (accX, accA) that the SIMD can interleave.
// ---------------------------------------------------------------------------
template <bool HAS_AGG, bool RES>
__global__ __launch_bounds__(256) void sage_layer_kernel(
    const float* __restrict__ X,            // [N,128] f32
    const float* __restrict__ AGG,          // [N,128] f32 (unused if !HAS_AGG)
    const float* __restrict__ deg,          // [N]     (unused if !HAS_AGG)
    const unsigned short* __restrict__ WpR, // packed root weight
    const unsigned short* __restrict__ WpL, // packed agg weight
    const float* __restrict__ bias,
    const float* __restrict__ bng, const float* __restrict__ bnb,
    const float* __restrict__ bnm, const float* __restrict__ bnv,
    float* __restrict__ Hout) {
    __shared__ unsigned short sX[16 * LDST];
    __shared__ unsigned short sA[HAS_AGG ? 16 * LDST : 16];

    const int node0 = blockIdx.x * 16;
    const int tid   = threadIdx.x;

    // ---- stage 16x128 f32 tiles -> bf16 LDS (8 cols per thread) ----
    {
        int row  = tid >> 4;            // 0..15
        int colb = (tid & 15) * 8;      // 0..120
        const float* px = X + (size_t)(node0 + row) * GH + colb;
        float4 a = *(const float4*)(px);
        float4 b = *(const float4*)(px + 4);
        unsigned short* d = sX + row * LDST + colb;
        d[0] = f2bf(a.x); d[1] = f2bf(a.y); d[2] = f2bf(a.z); d[3] = f2bf(a.w);
        d[4] = f2bf(b.x); d[5] = f2bf(b.y); d[6] = f2bf(b.z); d[7] = f2bf(b.w);
        if (HAS_AGG) {
            float inv = 1.0f / fmaxf(deg[node0 + row], 1.0f);  // segment mean
            const float* pa = AGG + (size_t)(node0 + row) * GH + colb;
            float4 c = *(const float4*)(pa);
            float4 e = *(const float4*)(pa + 4);
            unsigned short* da = sA + row * LDST + colb;
            da[0] = f2bf(c.x * inv); da[1] = f2bf(c.y * inv);
            da[2] = f2bf(c.z * inv); da[3] = f2bf(c.w * inv);
            da[4] = f2bf(e.x * inv); da[5] = f2bf(e.y * inv);
            da[6] = f2bf(e.z * inv); da[7] = f2bf(e.w * inv);
        }
    }
    __syncthreads();

    const int lane = tid & 31;
    const int nt   = tid >> 5;          // wave id == N-tile
    v8f accX = {};
    v8f accA = {};
#pragma unroll
    for (int kt = 0; kt < 4; ++kt) {
        v16bf aX = load_a_frag(sX, lane, kt);
        v16bf bR = *(const v16bf*)(WpR + ((size_t)(kt * 8 + nt) * 32 + lane) * 16);
        accX = __builtin_amdgcn_wmma_f32_16x16x32_bf16(
            false, aX, false, bR, (short)0, accX, false, false);
        if (HAS_AGG) {
            v16bf aA = load_a_frag(sA, lane, kt);
            v16bf bL = *(const v16bf*)(WpL + ((size_t)(kt * 8 + nt) * 32 + lane) * 16);
            accA = __builtin_amdgcn_wmma_f32_16x16x32_bf16(
                false, aA, false, bL, (short)0, accA, false, false);
        }
    }

    // ---- epilogue: bias + folded BN + ReLU + residual, in place ----
    const int col   = nt * 16 + (lane & 15);
    const int rbase = (lane >> 4) * 8;
    float scale = bng[col] * rsqrtf(bnv[col] + 1e-5f);
    float shift = bnb[col] - bnm[col] * scale;
    float bi    = bias[col];
#pragma unroll
    for (int i = 0; i < 8; ++i) {
        int row = rbase + i;
        float s = HAS_AGG ? (accX[i] + accA[i]) : accX[i];
        float t = (s + bi) * scale + shift;
        t = fmaxf(t, 0.0f);
        size_t idx = (size_t)(node0 + row) * GH + col;
        if (RES) t += X[idx];
        Hout[idx] = t;
    }
}

// ---------------------------------------------------------------------------
// Output head: out = h @ out_W + out_b  (47 cols -> 3 padded N-tiles).
// Block = 128 thr = 4 waves; waves 0..2 own the 3 column tiles.
// ---------------------------------------------------------------------------
__global__ __launch_bounds__(128) void out_kernel(
    const float* __restrict__ H, const unsigned short* __restrict__ WpO,
    const float* __restrict__ ob, float* __restrict__ out) {
    __shared__ unsigned short sX[16 * LDST];
    const int node0 = blockIdx.x * 16;
    const int tid   = threadIdx.x;
    {
        int row  = tid >> 3;            // 0..15
        int colb = (tid & 7) * 16;      // 0..112
        const float* px = H + (size_t)(node0 + row) * GH + colb;
        unsigned short* d = sX + row * LDST + colb;
#pragma unroll
        for (int j = 0; j < 4; ++j) {
            float4 a = *(const float4*)(px + j * 4);
            d[j * 4 + 0] = f2bf(a.x); d[j * 4 + 1] = f2bf(a.y);
            d[j * 4 + 2] = f2bf(a.z); d[j * 4 + 3] = f2bf(a.w);
        }
    }
    __syncthreads();

    const int lane = tid & 31;
    const int nt   = tid >> 5;
    if (nt >= 3) return;
    v8f acc = {};
#pragma unroll
    for (int kt = 0; kt < 4; ++kt) {
        v16bf a = load_a_frag(sX, lane, kt);
        v16bf b = *(const v16bf*)(WpO + ((size_t)(kt * 3 + nt) * 32 + lane) * 16);
        acc = __builtin_amdgcn_wmma_f32_16x16x32_bf16(
            false, a, false, b, (short)0, acc, false, false);
    }
    const int col = nt * 16 + (lane & 15);
    if (col < GC) {
        const int rbase = (lane >> 4) * 8;
        float bi = ob[col];
#pragma unroll
        for (int i = 0; i < 8; ++i)
            out[(size_t)(node0 + rbase + i) * GC + col] = acc[i] + bi;
    }
}

// ---------------------------------------------------------------------------
// Host-side orchestration (graph-capture safe: only kernels + memsetAsync).
// ---------------------------------------------------------------------------
extern "C" void kernel_launch(void* const* d_in, const int* in_sizes, int n_in,
                              void* d_out, int out_size, void* d_ws, size_t ws_size,
                              hipStream_t stream) {
    const float* x     = (const float*)d_in[0];
    const float* in_W  = (const float*)d_in[1];
    const float* in_b  = (const float*)d_in[2];
    const float* ibn_g = (const float*)d_in[3];
    const float* ibn_b = (const float*)d_in[4];
    const float* ibn_m = (const float*)d_in[5];
    const float* ibn_v = (const float*)d_in[6];
    const float* Wl    = (const float*)d_in[7];   // [3][128][128]
    const float* bl    = (const float*)d_in[8];   // [3][128]
    const float* Wr    = (const float*)d_in[9];   // [3][128][128]
    const float* bn_g  = (const float*)d_in[10];  // [3][128]
    const float* bn_b  = (const float*)d_in[11];
    const float* bn_m  = (const float*)d_in[12];
    const float* bn_v  = (const float*)d_in[13];
    const float* out_W = (const float*)d_in[14];  // [128][47]
    const float* out_b = (const float*)d_in[15];  // [47]
    const int*   ei    = (const int*)d_in[16];    // [2][E]
    const int*   src   = ei;
    const int*   dst   = ei + GE;

    // workspace layout (all offsets 32B-aligned)
    char* ws = (char*)d_ws;
    const size_t HB = (size_t)GN * GH * sizeof(float);        // 51.2 MB
    float*          h    = (float*)(ws);
    float*          agg  = (float*)(ws + HB);
    float*          deg  = (float*)(ws + 2 * HB);
    unsigned short* wp   = (unsigned short*)(ws + 2 * HB + (size_t)GN * sizeof(float));
    unsigned short* wpIn = wp;                         // 16384 each
    unsigned short* wpL0 = wp + 16384 * 1;
    unsigned short* wpR0 = wp + 16384 * 4;
    unsigned short* wpO  = wp + 16384 * 7;             // 6144

    // 1) pack weights into WMMA B-fragment layout (bf16)
    pack128_kernel<<<64, 256, 0, stream>>>(in_W, wpIn);
    for (int l = 0; l < GL; ++l) {
        pack128_kernel<<<64, 256, 0, stream>>>(Wl + (size_t)l * GH * GH, wpL0 + 16384 * l);
        pack128_kernel<<<64, 256, 0, stream>>>(Wr + (size_t)l * GH * GH, wpR0 + 16384 * l);
    }
    packout_kernel<<<24, 256, 0, stream>>>(out_W, wpO);

    // 2) degrees
    hipMemsetAsync(deg, 0, (size_t)GN * sizeof(float), stream);
    degree_kernel<<<(GE + 255) / 256, 256, 0, stream>>>(dst, deg, GE);

    // 3) input layer: h = relu(bn(x @ in_W + in_b))
    sage_layer_kernel<false, false><<<GN / 16, 256, 0, stream>>>(
        x, nullptr, nullptr, wpIn, nullptr, in_b,
        ibn_g, ibn_b, ibn_m, ibn_v, h);

    // 4) three SAGE layers with residual
    for (int l = 0; l < GL; ++l) {
        hipMemsetAsync(agg, 0, HB, stream);
        scatter_kernel<<<GE / 8, 256, 0, stream>>>(h, src, dst, agg, GE);
        sage_layer_kernel<true, true><<<GN / 16, 256, 0, stream>>>(
            h, agg, deg,
            wpR0 + 16384 * l, wpL0 + 16384 * l, bl + (size_t)l * GH,
            bn_g + (size_t)l * GH, bn_b + (size_t)l * GH,
            bn_m + (size_t)l * GH, bn_v + (size_t)l * GH,
            h);
    }

    // 5) classifier head
    out_kernel<<<GN / 16, 128, 0, stream>>>(h, wpO, out_b, (float*)d_out);
}
